// GATEdgeClassifier_13829794693457
// MI455X (gfx1250) — matrix operand order repaired and checked by reading, not compiled, not run on hardware
//
#include <hip/hip_runtime.h>
#include <stdint.h>

// ---------------- problem constants ----------------
#define NN     20000          // nodes
#define EE     320000         // edges
#define ETOT   (EE + NN)      // edges + self loops
#define FIN    128            // in_channels
#define HID    64
#define HEADS  4
#define HC     (HEADS * HID)  // 256
#define NEG_SLOPE 0.2f

// ---------------- WMMA types ----------------
typedef __bf16 bf16x16 __attribute__((ext_vector_type(16)));
typedef float  f32x8   __attribute__((ext_vector_type(8)));
union Frag { bf16x16 v; uint32_t u[8]; };

__device__ __forceinline__ unsigned short f2bf(float f) {
    unsigned int u = __float_as_uint(f);
    u = (u + 0x7FFFu + ((u >> 16) & 1u)) >> 16;   // RNE
    return (unsigned short)u;
}
__device__ __forceinline__ float bf2f(unsigned short h) {
    return __uint_as_float(((unsigned int)h) << 16);
}
__device__ __forceinline__ void atomicMaxF(float* addr, float val) {
    unsigned int* ua = (unsigned int*)addr;
    unsigned int old = *ua;
    while (__uint_as_float(old) < val) {
        unsigned int assumed = old;
        old = atomicCAS(ua, assumed, __float_as_uint(val));
        if (old == assumed) break;
    }
}

// ---------------- casts ----------------
__global__ void k_cast_bf16(const float* __restrict__ src,
                            unsigned short* __restrict__ dst, int n) {
    int i = blockIdx.x * blockDim.x + threadIdx.x;
    if (i < n) dst[i] = f2bf(src[i]);
}

// W [K, Nout] row-major fp32  ->  Wt [Nout, K] row-major bf16
__global__ void k_transpose_cast(const float* __restrict__ W,
                                 unsigned short* __restrict__ Wt,
                                 int K, int Nout) {
    int i = blockIdx.x * blockDim.x + threadIdx.x;
    if (i >= K * Nout) return;
    int n = i / K, k = i % K;
    Wt[i] = f2bf(W[k * Nout + n]);
}

// ---------------- generic WMMA GEMM ----------------
// C[M,Nout] = A[M,K] (bf16 rm) @ Bt[Nout,K]^T (bf16 rm, i.e. B[K,Nout])
// M % 16 == 0, K % 32 == 0 (K <= 128), Nout % 64 == 0.
// Block: 8 waves. blockIdx.y selects a 64-column group of B, staged in LDS
// once per block. Each wave owns one M-tile and computes 4 N-tiles.
// A sched_barrier splits each k-step into [all loads] -> [4 WMMAs] so the
// 4 B fragments stay simultaneously live (distinct VGPRs, one dscnt wait).
__global__ void k_gemm_bf16(const unsigned short* __restrict__ A,
                            const unsigned short* __restrict__ Bt,
                            float* __restrict__ C,
                            int M, int K, int Nout) {
    __shared__ unsigned short sB[64 * 128];   // 64 cols x K (K<=128), 16 KB

    int tid = threadIdx.x;
    int ng  = blockIdx.y;                      // 64-column group
    // cooperative stage of B columns [ng*64, ng*64+64) into LDS
    for (int i = tid; i < 64 * K; i += 256) {
        int col = i / K, k = i % K;
        sB[col * K + k] = Bt[(ng * 64 + col) * K + k];
    }
    __syncthreads();

    int wave = tid >> 5;
    int lane = tid & 31;
    int half = lane >> 4;
    int lr   = lane & 15;
    int mt   = blockIdx.x * 8 + wave;          // M-tile owned by this wave
    if (mt * 16 >= M) return;                  // after the barrier: safe

    int row = mt * 16 + lr;                    // A row for this lane

    f32x8 acc[4] = {};
    for (int kk = 0; kk < K; kk += 32) {
        Frag a;
        const uint32_t* arow = (const uint32_t*)(A + row * K + kk);
        #pragma unroll
        for (int v = 0; v < 4; ++v) {
            a.u[v]     = arow[v + 4 * half];        // K = 2v    + 8*half
            a.u[v + 4] = arow[v + 8 + 4 * half];    // K = 2v+16 + 8*half
        }
        Frag b[4];                                  // all 4 live at once
        #pragma unroll
        for (int nt = 0; nt < 4; ++nt) {
            int lcol = nt * 16 + lr;                // column within LDS group
            const uint32_t* brow = (const uint32_t*)(sB + lcol * K + kk);
            #pragma unroll
            for (int v = 0; v < 8; ++v) b[nt].u[v] = brow[(v & 3) + 4 * half + 8 * (v >> 2)];
        }
        __builtin_amdgcn_sched_barrier(0);          // loads above, WMMAs below
        #pragma unroll
        for (int nt = 0; nt < 4; ++nt)
            acc[nt] = __builtin_amdgcn_wmma_f32_16x16x32_bf16(
                          false, a.v, false, b[nt].v, (short)0, acc[nt], false, false);
    }
    #pragma unroll
    for (int nt = 0; nt < 4; ++nt) {
        int ncol = ng * 64 + nt * 16 + lr;
        #pragma unroll
        for (int r = 0; r < 8; ++r) {
            int m = mt * 16 + r + 8 * half;
            C[m * Nout + ncol] = acc[nt][r];
        }
    }
}

// ---------------- attention plumbing ----------------
__global__ void k_scores(const float* __restrict__ H,      // [N, 256]
                         const float* __restrict__ asrc,   // [4,64]
                         const float* __restrict__ adst,
                         float* __restrict__ ssrc, float* __restrict__ sdst) {
    int idx = blockIdx.x * blockDim.x + threadIdx.x;
    if (idx >= NN * HEADS) return;
    int n = idx >> 2, h = idx & 3;
    const float* hp = H + n * HC + h * HID;
    float a = 0.f, b = 0.f;
    for (int c = 0; c < HID; ++c) {
        float v = hp[c];
        a += v * asrc[h * HID + c];
        b += v * adst[h * HID + c];
    }
    ssrc[idx] = a; sdst[idx] = b;
}

__global__ void k_init(float* __restrict__ agg, float* __restrict__ m,
                       float* __restrict__ z) {
    int idx = blockIdx.x * blockDim.x + threadIdx.x;
    if (idx < NN * HID) agg[idx] = 0.f;
    if (idx < NN * HEADS) { m[idx] = -3.0e38f; z[idx] = 0.f; }
}

__global__ void k_edge_score(const int* __restrict__ ei,
                             const float* __restrict__ ssrc,
                             const float* __restrict__ sdst,
                             float* __restrict__ e, float* __restrict__ m) {
    int et = blockIdx.x * blockDim.x + threadIdx.x;
    if (et >= ETOT) return;
    int s = (et < EE) ? ei[et]      : (et - EE);
    int d = (et < EE) ? ei[EE + et] : (et - EE);
    #pragma unroll
    for (int h = 0; h < HEADS; ++h) {
        float v = ssrc[s * HEADS + h] + sdst[d * HEADS + h];
        v = (v > 0.f) ? v : NEG_SLOPE * v;
        e[et * HEADS + h] = v;
        atomicMaxF(&m[d * HEADS + h], v);
    }
}

__global__ void k_edge_exp(const int* __restrict__ ei,
                           float* __restrict__ e,        // in: e, out: p
                           const float* __restrict__ m,
                           float* __restrict__ z) {
    int et = blockIdx.x * blockDim.x + threadIdx.x;
    if (et >= ETOT) return;
    int d = (et < EE) ? ei[EE + et] : (et - EE);
    #pragma unroll
    for (int h = 0; h < HEADS; ++h) {
        float p = __expf(e[et * HEADS + h] - m[d * HEADS + h]);
        e[et * HEADS + h] = p;
        atomicAdd(&z[d * HEADS + h], p);
    }
}

// one block (64 threads) per edge: agg[d] += mean_h( alpha_h * h[src][h] )
__global__ void k_edge_agg(const int* __restrict__ ei,
                           const float* __restrict__ p,
                           const float* __restrict__ z,
                           const float* __restrict__ H,   // [N,256]
                           float* __restrict__ agg) {
    int et = blockIdx.x;
    int c  = threadIdx.x;                  // 0..63
    int s = (et < EE) ? ei[et]      : (et - EE);
    int d = (et < EE) ? ei[EE + et] : (et - EE);
    float acc = 0.f;
    #pragma unroll
    for (int h = 0; h < HEADS; ++h) {
        float alpha = p[et * HEADS + h] / z[d * HEADS + h];
        acc += alpha * H[s * HC + h * HID + c];
    }
    atomicAdd(&agg[d * HID + c], 0.25f * acc);
}

// out = elu(agg + gat_b + res (+ res_b)); also emit bf16 copy for next GEMM
__global__ void k_finalize(const float* __restrict__ agg,
                           const float* __restrict__ gb,
                           const float* __restrict__ res,
                           const float* __restrict__ resb,   // nullptr if i>0
                           float* __restrict__ hout,
                           unsigned short* __restrict__ hb) {
    int idx = blockIdx.x * blockDim.x + threadIdx.x;
    if (idx >= NN * HID) return;
    int c = idx & 63;
    float v = agg[idx] + gb[c] + res[idx];
    if (resb) v += resb[c];
    v = (v > 0.f) ? v : (__expf(v) - 1.f);
    hout[idx] = v;
    hb[idx]   = f2bf(v);
}

// ---------------- fused edge MLP ----------------
// er = concat(h[row], h[col])  (gathered, K=128)
// t1 = relu(er@W0 + b0) ; t2 = relu(t1@W1 + b1 + t1) ; logit = t2@w2 + b2
// 8 waves/block, 16 edges/wave -> 128 edges/block. E % 128 == 0.
__global__ void k_edge_mlp(const int* __restrict__ ei,
                           const unsigned short* __restrict__ hb,   // [N,64] bf16
                           const unsigned short* __restrict__ W0t,  // [64,128] bf16
                           const unsigned short* __restrict__ W1t,  // [64,64]  bf16
                           const float* __restrict__ b0,
                           const float* __restrict__ b1,
                           const float* __restrict__ w2,            // [64]
                           const float* __restrict__ b2,            // [1]
                           float* __restrict__ out) {
    __shared__ unsigned short sW0[64 * 128];
    __shared__ unsigned short sW1[64 * 64];
    __shared__ float sb0[64], sb1[64], sw2[64];
    __shared__ float sb2;
    __shared__ unsigned short tile[8][16][64];   // per-wave activation tile

    int tid = threadIdx.x;
    for (int i = tid; i < 64 * 128; i += 256) sW0[i] = W0t[i];
    for (int i = tid; i < 64 * 64;  i += 256) sW1[i] = W1t[i];
    if (tid < 64) { sb0[tid] = b0[tid]; sb1[tid] = b1[tid]; sw2[tid] = w2[tid]; }
    if (tid == 0) sb2 = b2[0];
    __syncthreads();

    int wave = tid >> 5, lane = tid & 31;
    int half = lane >> 4, lr = lane & 15;
    int ebase  = blockIdx.x * 128 + wave * 16;
    int myEdge = ebase + lr;
    int nr = ei[myEdge];
    int nc = ei[EE + myEdge];
    const uint32_t* hr = (const uint32_t*)(hb + nr * HID);
    const uint32_t* hc = (const uint32_t*)(hb + nc * HID);

    // -------- layer 1: [16,128] x [128,64] --------
    f32x8 acc[4] = {};
    #pragma unroll
    for (int kk = 0; kk < 128; kk += 32) {
        Frag a;
        #pragma unroll
        for (int v = 0; v < 4; ++v) {
            int k0 = kk + 2 * v + 8 * half;
            int k1 = kk + 2 * v + 16 + 8 * half;
            a.u[v]     = (k0 < 64) ? hr[k0 >> 1] : hc[(k0 - 64) >> 1];
            a.u[v + 4] = (k1 < 64) ? hr[k1 >> 1] : hc[(k1 - 64) >> 1];
        }
        Frag b[4];
        #pragma unroll
        for (int nt = 0; nt < 4; ++nt) {
            int ncol = nt * 16 + lr;
            const uint32_t* brow = (const uint32_t*)(sW0 + ncol * 128 + kk);
            #pragma unroll
            for (int v = 0; v < 8; ++v) b[nt].u[v] = brow[(v & 3) + 4 * half + 8 * (v >> 2)];
        }
        __builtin_amdgcn_sched_barrier(0);
        #pragma unroll
        for (int nt = 0; nt < 4; ++nt)
            acc[nt] = __builtin_amdgcn_wmma_f32_16x16x32_bf16(
                          false, a.v, false, b[nt].v, (short)0, acc[nt], false, false);
    }
    #pragma unroll
    for (int nt = 0; nt < 4; ++nt) {
        int ncol = nt * 16 + lr;
        #pragma unroll
        for (int r = 0; r < 8; ++r) {
            int m = r + 8 * half;
            float v = acc[nt][r] + sb0[ncol];
            v = (v > 0.f) ? v : 0.f;
            tile[wave][m][ncol] = f2bf(v);
        }
    }
    __syncthreads();

    // -------- layer 2: [16,64] x [64,64] + residual --------
    f32x8 acc2[4] = {};
    const uint32_t* trow = (const uint32_t*)&tile[wave][lr][0];
    #pragma unroll
    for (int kk = 0; kk < 64; kk += 32) {
        Frag a;
        #pragma unroll
        for (int v = 0; v < 4; ++v) {
            a.u[v]     = trow[(kk >> 1) + v + 4 * half];
            a.u[v + 4] = trow[(kk >> 1) + v + 8 + 4 * half];
        }
        Frag b[4];
        #pragma unroll
        for (int nt = 0; nt < 4; ++nt) {
            int ncol = nt * 16 + lr;
            const uint32_t* brow = (const uint32_t*)(sW1 + ncol * 64 + kk);
            #pragma unroll
            for (int v = 0; v < 8; ++v) b[nt].u[v] = brow[(v & 3) + 4 * half + 8 * (v >> 2)];
        }
        __builtin_amdgcn_sched_barrier(0);
        #pragma unroll
        for (int nt = 0; nt < 4; ++nt)
            acc2[nt] = __builtin_amdgcn_wmma_f32_16x16x32_bf16(
                           false, a.v, false, b[nt].v, (short)0, acc2[nt], false, false);
    }
    __syncthreads();   // all A reads of tile done before overwriting it
    #pragma unroll
    for (int nt = 0; nt < 4; ++nt) {
        int ncol = nt * 16 + lr;
        #pragma unroll
        for (int r = 0; r < 8; ++r) {
            int m = r + 8 * half;
            float t1v = bf2f(tile[wave][m][ncol]);
            float v = acc2[nt][r] + sb1[ncol] + t1v;
            v = (v > 0.f) ? v : 0.f;
            tile[wave][m][ncol] = f2bf(v);
        }
    }
    __syncthreads();

    // -------- layer 3: [16,64] x [64,1] --------
    if (lane < 16) {
        float s = sb2;
        for (int c = 0; c < 64; ++c) s += bf2f(tile[wave][lane][c]) * sw2[c];
        out[ebase + lane] = s;
    }
}

// ---------------- host launch ----------------
extern "C" void kernel_launch(void* const* d_in, const int* in_sizes, int n_in,
                              void* d_out, int out_size, void* d_ws, size_t ws_size,
                              hipStream_t stream) {
    const float* x      = (const float*)d_in[0];
    const int*   ei     = (const int*)  d_in[1];
    const float* W[4]   = { (const float*)d_in[2], (const float*)d_in[3],
                            (const float*)d_in[4], (const float*)d_in[5] };
    const float* a_src  = (const float*)d_in[6];
    const float* a_dst  = (const float*)d_in[7];
    const float* gat_b  = (const float*)d_in[8];
    const float* res_W  = (const float*)d_in[9];
    const float* res_b  = (const float*)d_in[10];
    const float* lin_W0 = (const float*)d_in[11];
    const float* lin_b0 = (const float*)d_in[12];
    const float* lin_W1 = (const float*)d_in[13];
    const float* lin_b1 = (const float*)d_in[14];
    const float* lin_W2 = (const float*)d_in[15];
    const float* lin_b2 = (const float*)d_in[16];
    float* out = (float*)d_out;

    char* base = (char*)d_ws;
    size_t off = 0;
    auto alloc = [&](size_t bytes) -> char* {
        char* p = base + off;
        off = (off + bytes + 255) & ~(size_t)255;
        return p;
    };
    unsigned short* xb    = (unsigned short*)alloc((size_t)NN * FIN * 2);
    unsigned short* Wt0   = (unsigned short*)alloc(256 * 128 * 2);
    unsigned short* Wt1   = (unsigned short*)alloc(256 * 64 * 2);
    unsigned short* Wt2   = (unsigned short*)alloc(256 * 64 * 2);
    unsigned short* Wt3   = (unsigned short*)alloc(256 * 64 * 2);
    unsigned short* resWt = (unsigned short*)alloc(64 * 128 * 2);
    unsigned short* lW0t  = (unsigned short*)alloc(64 * 128 * 2);
    unsigned short* lW1t  = (unsigned short*)alloc(64 * 64 * 2);
    float* resid = (float*)alloc((size_t)NN * HID * 4);
    float* Hh    = (float*)alloc((size_t)NN * HC * 4);
    float* ssrc  = (float*)alloc((size_t)NN * HEADS * 4);
    float* sdst  = (float*)alloc((size_t)NN * HEADS * 4);
    float* mbuf  = (float*)alloc((size_t)NN * HEADS * 4);
    float* zbuf  = (float*)alloc((size_t)NN * HEADS * 4);
    float* ebuf  = (float*)alloc((size_t)ETOT * HEADS * 4);
    float* agg   = (float*)alloc((size_t)NN * HID * 4);
    float* hA    = (float*)alloc((size_t)NN * HID * 4);
    float* hB    = (float*)alloc((size_t)NN * HID * 4);
    unsigned short* hb16 = (unsigned short*)alloc((size_t)NN * HID * 2);

    unsigned short* Wt[4] = { Wt0, Wt1, Wt2, Wt3 };

    const int MT_BLOCKS = (NN / 16 + 7) / 8;   // 157 blocks of 8 M-tile waves

    // casts / transposes
    k_cast_bf16<<<(NN * FIN + 255) / 256, 256, 0, stream>>>(x, xb, NN * FIN);
    k_transpose_cast<<<(128 * 256 + 255) / 256, 256, 0, stream>>>(W[0], Wt0, 128, 256);
    for (int i = 1; i < 4; ++i)
        k_transpose_cast<<<(64 * 256 + 255) / 256, 256, 0, stream>>>(W[i], Wt[i], 64, 256);
    k_transpose_cast<<<(128 * 64 + 255) / 256, 256, 0, stream>>>(res_W,  resWt, 128, 64);
    k_transpose_cast<<<(128 * 64 + 255) / 256, 256, 0, stream>>>(lin_W0, lW0t,  128, 64);
    k_transpose_cast<<<(64 * 64 + 255) / 256, 256, 0, stream>>>(lin_W1, lW1t,  64, 64);

    // residual projection: x @ res_W  -> resid [N,64]
    k_gemm_bf16<<<dim3(MT_BLOCKS, 1), 256, 0, stream>>>(xb, resWt, resid, NN, 128, 64);

    float* hCur = hA;   // holds layer output (fp32)
    float* hNxt = hB;
    for (int i = 0; i < 4; ++i) {
        const unsigned short* A = (i == 0) ? xb : hb16;
        int K = (i == 0) ? 128 : 64;
        k_gemm_bf16<<<dim3(MT_BLOCKS, 4), 256, 0, stream>>>(A, Wt[i], Hh, NN, K, HC);
        k_scores<<<(NN * HEADS + 255) / 256, 256, 0, stream>>>(
            Hh, a_src + i * HC, a_dst + i * HC, ssrc, sdst);
        k_init<<<(NN * HID + 255) / 256, 256, 0, stream>>>(agg, mbuf, zbuf);
        k_edge_score<<<(ETOT + 255) / 256, 256, 0, stream>>>(ei, ssrc, sdst, ebuf, mbuf);
        k_edge_exp<<<(ETOT + 255) / 256, 256, 0, stream>>>(ei, ebuf, mbuf, zbuf);
        k_edge_agg<<<ETOT, 64, 0, stream>>>(ei, ebuf, zbuf, Hh, agg);
        const float* res  = (i == 0) ? resid : hCur;
        const float* resb = (i == 0) ? res_b : nullptr;
        k_finalize<<<(NN * HID + 255) / 256, 256, 0, stream>>>(
            agg, gat_b + i * HID, res, resb, hNxt, hb16);
        float* t = hCur; hCur = hNxt; hNxt = t;
    }

    // fused edge classifier MLP (reads hb16 = bf16 of final node features)
    k_edge_mlp<<<EE / 128, 256, 0, stream>>>(
        ei, hb16, lW0t, lW1t, lin_b0, lin_b1, lin_W2, lin_b2, out);

    (void)in_sizes; (void)n_in; (void)out_size; (void)ws_size;
}
